// GNN_Model_19464791786105
// MI455X (gfx1250) — compile-verified, matrix operand order checked by reading, not run on hardware
//
#include <hip/hip_runtime.h>
#include <hip/hip_bf16.h>
#include <math.h>

// ---------------------------------------------------------------------------
// GNN pipeline for MI455X (gfx1250, wave32).
//  - Skinny M=8 GEMMs (pnode 402MB weights, d1 104MB weights): HBM-bound,
//    VALU FMA, weights read exactly once (roofline floor ~22us @ 23.3TB/s).
//  - 128-wide SAGE GEMMs + conv1d implicit GEMMs: WMMA bf16.  B operands are
//    prepacked into fragment order, async-copied to LDS once per block
//    (global_load_async_to_lds_b128 / s_wait_asynccnt), and fed to the XDL
//    pipe as back-to-back wmma chains.  A operands: 4x b128 loads + cvt.
//  - Graph segment-means: returnless global_atomic_add_f32 (L2-resident).
// ---------------------------------------------------------------------------

#define E_EDGES 1000000
#define NE_HALF 500000
#define PN_ROWS 32768
#define FN_ROWS 6144

typedef __attribute__((ext_vector_type(16))) __bf16 v16bf;
typedef __attribute__((ext_vector_type(8)))  float  v8f;

__device__ __forceinline__ __bf16 f2bf(float f) {
  union { float f; unsigned u; } v; v.f = f;
  unsigned r = v.u + 0x7FFFu + ((v.u >> 16) & 1u);   // round-to-nearest-even
  unsigned short h = (unsigned short)(r >> 16);
  __bf16 o; __builtin_memcpy(&o, &h, 2);
  return o;
}

// K index of element e (0..15) of a 16-bit A/B fragment, wave32 layout:
// VGPR0-3: K = hi*8 + 0..7 ; VGPR4-7: K = 16 + hi*8 + 0..7
__device__ __forceinline__ int kpat(int e, int hi) {
  return (e < 8) ? (hi * 8 + e) : (16 + hi * 8 + (e - 8));
}

// CDNA5 async global->LDS copy (ASYNCcnt-tracked); lds_off = low 32 bits of
// the generic address of a __shared__ object (== workgroup LDS byte address).
__device__ __forceinline__ void async_copy_b128(unsigned lds_off, const void* g) {
  asm volatile("global_load_async_to_lds_b128 %0, %1, off"
               :: "v"(lds_off), "v"((unsigned long long)(size_t)g) : "memory");
}
__device__ __forceinline__ void wait_async0() {
  asm volatile("s_wait_asynccnt 0x0" ::: "memory");
}

// ---------------------------------------------------------------- utilities
__global__ void zero_kernel(float* __restrict__ p, int n) {
  int i = blockIdx.x * blockDim.x + threadIdx.x;
  if (i < n) p[i] = 0.f;
}

__global__ void count_edges(const int* __restrict__ ei, float* __restrict__ cnt,
                            int nE, int fwd) {
  int e = blockIdx.x * blockDim.x + threadIdx.x;
  if (e >= nE) return;
  int dst = fwd ? ei[E_EDGES + 2 * e] : ei[2 * e + 1];
  atomicAdd(&cnt[dst], 1.0f);
}

// agg[dst,:d] += xsrc[s,:d]   (row stride 128)
__global__ void scatter_add(const int* __restrict__ ei, const float* __restrict__ xsrc,
                            float* __restrict__ agg, int nE, int d, int fwd) {
  int i = blockIdx.x * blockDim.x + threadIdx.x;
  if (i >= nE * d) return;
  int e = i / d, c = i % d;
  int s, dst;
  if (fwd) { s = ei[2 * e];               dst = ei[E_EDGES + 2 * e]; }
  else     { s = ei[E_EDGES + 2 * e + 1]; dst = ei[2 * e + 1]; }
  atomicAdd(&agg[dst * 128 + c], xsrc[s * 128 + c]);
}

// Pack a (K x N) f32 weight into WMMA-B fragment order, bf16, zero-padded to
// Kr=ceil32(K).  Flat idx = ((ks*NT + nt)*32 + lane)*16 + e.
// Value = W[k*sk + n*sn], k = ks*32 + kpat(e, lane>>4), n = nt*16 + lane&15.
__global__ void pack_wmma_b(const float* __restrict__ W, __bf16* __restrict__ P,
                            int K, int NT, int sk, int sn, int total) {
  int idx = blockIdx.x * blockDim.x + threadIdx.x;
  if (idx >= total) return;
  int e = idx & 15, lane = (idx >> 4) & 31;
  int rest = idx >> 9;
  int nt = rest % NT, ks = rest / NT;
  int k = ks * 32 + kpat(e, lane >> 4);
  int n = nt * 16 + (lane & 15);
  float v = (k < K) ? W[k * sk + n * sn] : 0.f;
  P[idx] = f2bf(v);
}

// ------------------------------------------------------- node encoders
__global__ void __launch_bounds__(128)
pnode_gemm(const float* __restrict__ xp, const float* __restrict__ W,
           const float* __restrict__ bias, float* __restrict__ out) {
  __shared__ float xt[8 * 1024];
  int n = blockIdx.x * 128 + threadIdx.x;
  float acc[8];
  for (int b = 0; b < 8; ++b) acc[b] = 0.f;
  for (int k0 = 0; k0 < 8192; k0 += 1024) {
    for (int j = threadIdx.x; j < 8 * 1024; j += 128) {
      int b = j >> 10, kk = j & 1023;
      xt[j] = xp[b * 8192 + k0 + kk];
    }
    __syncthreads();
    for (int kk = 0; kk < 1024; ++kk) {
      float w = W[(long long)(k0 + kk) * 12288 + n];
#pragma unroll
      for (int b = 0; b < 8; ++b) acc[b] = fmaf(xt[b * 1024 + kk], w, acc[b]);
    }
    __syncthreads();
  }
  for (int b = 0; b < 8; ++b) {
    float v = acc[b] + bias[n];
    int flat = b * 12288 + n;
    out[(flat / 3) * 128 + (flat % 3)] = v;
  }
}

__global__ void fnode_gemm(const float* __restrict__ xf, const float* __restrict__ W,
                           const float* __restrict__ bias, float* __restrict__ out) {
  int i = blockIdx.x * blockDim.x + threadIdx.x;
  if (i >= 2048 * 192) return;
  int r = i / 192, n = i % 192;
  float s = bias[n];
  for (int k = 0; k < 64; ++k) s = fmaf(xf[r * 64 + k], W[k * 192 + n], s);
  int flat = r * 192 + n;
  out[(flat / 3) * 128 + (flat % 3)] = s;
}

// ------------------------------------------------------- fused SAGE update
// out[r,0:128] = relu( (agg[r]/max(cnt,1)) @ wl + bl + xdst[r] @ wr )
// One wave per 16-row tile; grid is exact (nrows multiple of 128 per block).
// Both packed B images ((ksL+ksR)*8KB <= 64KB) are async-staged into LDS.
__global__ void __launch_bounds__(256)
sage_update(const float* __restrict__ agg, const float* __restrict__ cnt,
            const float* __restrict__ xdst,
            const v16bf* __restrict__ pwl, const float* __restrict__ bl,
            const v16bf* __restrict__ pwr,
            float* __restrict__ out, int ksL, int ksR) {
  __shared__ __align__(32) unsigned char bsh[65536];
  const int tid = threadIdx.x;
  const int wave = tid >> 5, lane = tid & 31;
  const int hi = lane >> 4, lo = lane & 15;
  const int r0 = (blockIdx.x * 8 + wave) * 16;

  // async-stage B images into LDS (16B granules, per-lane addresses)
  const unsigned base = (unsigned)(size_t)(void*)bsh;
  const int n16L = ksL * 512;                 // ksL*8 frag-groups *32 lanes *32B /16B
  const int n16R = ksR * 512;
  const char* gl = (const char*)pwl;
  const char* gr = (const char*)pwr;
  for (int i = tid; i < n16L; i += 256)
    async_copy_b128(base + (unsigned)i * 16u, gl + (size_t)i * 16);
  const unsigned offR = (unsigned)n16L * 16u;
  for (int i = tid; i < n16R; i += 256)
    async_copy_b128(base + offR + (unsigned)i * 16u, gr + (size_t)i * 16);
  wait_async0();
  __syncthreads();
  const v16bf* shL = (const v16bf*)bsh;
  const v16bf* shR = shL + ksL * 256;         // ksL*8*32 fragment slots

  const int arow = r0 + lo;
  const float rc = 1.0f / fmaxf(cnt[arow], 1.0f);

  v8f acc[8];
#pragma unroll
  for (int t = 0; t < 8; ++t)
#pragma unroll
    for (int j = 0; j < 8; ++j) acc[t][j] = 0.f;

#pragma unroll
  for (int phase = 0; phase < 2; ++phase) {
    const float* Arow = (phase ? xdst : agg) + arow * 128;
    const v16bf* shB = phase ? shR : shL;
    const int nks = phase ? ksR : ksL;
    const float sc = phase ? 1.0f : rc;
    for (int ks = 0; ks < nks; ++ks) {
      const float* ap = Arow + ks * 32 + hi * 8;
      float4 q0 = *(const float4*)(ap + 0);
      float4 q1 = *(const float4*)(ap + 4);
      float4 q2 = *(const float4*)(ap + 16);
      float4 q3 = *(const float4*)(ap + 20);
      v16bf a;
      a[0] = f2bf(q0.x * sc);  a[1] = f2bf(q0.y * sc);
      a[2] = f2bf(q0.z * sc);  a[3] = f2bf(q0.w * sc);
      a[4] = f2bf(q1.x * sc);  a[5] = f2bf(q1.y * sc);
      a[6] = f2bf(q1.z * sc);  a[7] = f2bf(q1.w * sc);
      a[8] = f2bf(q2.x * sc);  a[9] = f2bf(q2.y * sc);
      a[10] = f2bf(q2.z * sc); a[11] = f2bf(q2.w * sc);
      a[12] = f2bf(q3.x * sc); a[13] = f2bf(q3.y * sc);
      a[14] = f2bf(q3.z * sc); a[15] = f2bf(q3.w * sc);
      v16bf bfr[8];
#pragma unroll
      for (int nt = 0; nt < 8; ++nt) bfr[nt] = shB[(ks * 8 + nt) * 32 + lane];
#pragma unroll
      for (int nt = 0; nt < 8; ++nt)
        acc[nt] = __builtin_amdgcn_wmma_f32_16x16x32_bf16(
            false, a, false, bfr[nt], (short)0, acc[nt], false, false);
    }
  }
  // C/D layout: VGPR v -> M = v + hi*8, N = nt*16 + lo
#pragma unroll
  for (int nt = 0; nt < 8; ++nt)
#pragma unroll
    for (int v = 0; v < 8; ++v) {
      int row = r0 + v + hi * 8;
      out[row * 128 + nt * 16 + lo] = fmaxf(acc[nt][v] + bl[nt * 16 + lo], 0.f);
    }
}

// ------------------------------------------------------- LayerNorm (in place)
__global__ void ln_kernel(float* __restrict__ x, const float* __restrict__ w,
                          const float* __restrict__ b, int nrows) {
  int wave = threadIdx.x >> 5, lane = threadIdx.x & 31;
  int row = blockIdx.x * 8 + wave;
  if (row >= nrows) return;
  float v[4];
  float s = 0.f;
#pragma unroll
  for (int j = 0; j < 4; ++j) { v[j] = x[row * 128 + j * 32 + lane]; s += v[j]; }
  for (int off = 16; off > 0; off >>= 1) s += __shfl_xor(s, off, 32);
  float m = s * (1.f / 128.f);
  float q = 0.f;
#pragma unroll
  for (int j = 0; j < 4; ++j) { float d = v[j] - m; q += d * d; }
  for (int off = 16; off > 0; off >>= 1) q += __shfl_xor(q, off, 32);
  float rstd = rsqrtf(q * (1.f / 128.f) + 1e-5f);
#pragma unroll
  for (int j = 0; j < 4; ++j) {
    int c = j * 32 + lane;
    x[row * 128 + c] = (v[j] - m) * rstd * w[c] + b[c];
  }
}

// ------------------------------------------------------- conv1d as implicit GEMM
// in:(8,Ci,Lin)  pw: prepacked weights  out:(8,64,Lout)=relu(conv VALID).
// K = Ci*8; lane's two 8-value A runs are contiguous taps of ci0=ks*4+hi and
// ci0+2.  B staged through LDS in 64KB segments (16 k-steps).  Tail waves use
// clamped addresses; their WMMA rows are discarded at the guarded store.
__global__ void __launch_bounds__(256)
conv1d_relu_wmma(const float* __restrict__ in, const v16bf* __restrict__ pw,
                 const float* __restrict__ bias, float* __restrict__ out,
                 int Ci, int Lin, int Lout) {
  __shared__ __align__(32) unsigned char bsh[65536];
  const int tid = threadIdx.x;
  const int wave = tid >> 5, lane = tid & 31;
  const int hi = lane >> 4, lo = lane & 15;
  const int bb = blockIdx.y;
  const int t0 = (blockIdx.x * 8 + wave) * 16;
  const int tA = t0 + lo;
  const int tc = tA < Lout ? tA : Lout - 1;            // clamp, no branch
  const float* fbase = in + (long long)bb * Ci * Lin + tc;
  const int nks = Ci / 4;                              // (Ci*8)/32
  const unsigned base = (unsigned)(size_t)(void*)bsh;
  const v16bf* shB = (const v16bf*)bsh;

  v8f acc[4];
#pragma unroll
  for (int t = 0; t < 4; ++t)
#pragma unroll
    for (int j = 0; j < 8; ++j) acc[t][j] = 0.f;

  for (int ks0 = 0; ks0 < nks; ks0 += 16) {
    const int kcnt = (nks - ks0) < 16 ? (nks - ks0) : 16;
    const int n16 = kcnt * 256;                        // kcnt*4*32 slots *32B /16B
    const char* g = (const char*)(pw + (size_t)ks0 * 4 * 32);
    __syncthreads();                                   // protect LDS reuse (WAR)
    for (int i = tid; i < n16; i += 256)
      async_copy_b128(base + (unsigned)i * 16u, g + (size_t)i * 16);
    wait_async0();
    __syncthreads();
    for (int ks = ks0; ks < ks0 + kcnt; ++ks) {
      const int ci0 = ks * 4 + hi;
      const float* p0 = fbase + (long long)ci0 * Lin;
      const float* p1 = fbase + (long long)(ci0 + 2) * Lin;
      v16bf a;
#pragma unroll
      for (int e = 0; e < 8; ++e) {
        a[e] = f2bf(p0[e]);
        a[8 + e] = f2bf(p1[e]);
      }
      v16bf bfr[4];
#pragma unroll
      for (int nt = 0; nt < 4; ++nt) bfr[nt] = shB[((ks - ks0) * 4 + nt) * 32 + lane];
#pragma unroll
      for (int nt = 0; nt < 4; ++nt)
        acc[nt] = __builtin_amdgcn_wmma_f32_16x16x32_bf16(
            false, a, false, bfr[nt], (short)0, acc[nt], false, false);
    }
  }
#pragma unroll
  for (int nt = 0; nt < 4; ++nt)
#pragma unroll
    for (int v = 0; v < 8; ++v) {
      int t = t0 + v + hi * 8;
      if (t < Lout) {
        int n = nt * 16 + lo;
        out[(bb * 64 + n) * Lout + t] = fmaxf(acc[nt][v] + bias[n], 0.f);
      }
    }
}

// ------------------------------------------------------- d1: (8,260800)@(260800,100)
__global__ void __launch_bounds__(128)
d1_partial(const float* __restrict__ x, const float* __restrict__ w,
           float* __restrict__ out) {
  const int K = 260800, CH = 512;
  int n = threadIdx.x;
  int kb = blockIdx.x * CH;
  if (n >= 100) return;
  float acc[8];
#pragma unroll
  for (int b = 0; b < 8; ++b) acc[b] = 0.f;
  int ke = kb + CH < K ? kb + CH : K;
  for (int k = kb; k < ke; ++k) {
    float wv = w[(long long)k * 100 + n];
#pragma unroll
    for (int b = 0; b < 8; ++b) acc[b] = fmaf(x[b * 260800 + k], wv, acc[b]);
  }
#pragma unroll
  for (int b = 0; b < 8; ++b) atomicAdd(&out[b * 100 + n], acc[b]);
}

// ------------------------------------------------------- tail
__global__ void __launch_bounds__(256)
tail_kernel(const float* __restrict__ d1out, const float* __restrict__ d1b,
            const float* __restrict__ other,
            const float* __restrict__ d2w, const float* __restrict__ d2b,
            const float* __restrict__ d3w, const float* __restrict__ d3b,
            float* __restrict__ out) {
  __shared__ float h1[8 * 102];
  __shared__ float h2[8 * 200];
  __shared__ float lg[16];
  int tid = threadIdx.x;
  for (int i = tid; i < 8 * 102; i += 256) {
    int b = i / 102, c = i % 102;
    h1[i] = (c < 100) ? fmaxf(d1out[b * 100 + c] + d1b[c], 0.f)
                      : other[b * 2 + (c - 100)];
  }
  __syncthreads();
  for (int i = tid; i < 8 * 200; i += 256) {
    int b = i / 200, n = i % 200;
    float s = d2b[n];
    for (int k = 0; k < 102; ++k) s = fmaf(h1[b * 102 + k], d2w[k * 200 + n], s);
    h2[i] = fmaxf(s, 0.f);
  }
  __syncthreads();
  if (tid < 16) {
    int b = tid >> 1, n = tid & 1;
    float s = d3b[n];
    for (int k = 0; k < 200; ++k) s = fmaf(h2[b * 200 + k], d3w[k * 2 + n], s);
    lg[tid] = s;
  }
  __syncthreads();
  if (tid < 8) {
    float a = lg[2 * tid], c = lg[2 * tid + 1];
    float m = fmaxf(a, c);
    float ea = expf(a - m), ec = expf(c - m);
    float d = ea + ec;
    out[2 * tid] = ea / d;
    out[2 * tid + 1] = ec / d;
  }
}

// ---------------------------------------------------------------------------
extern "C" void kernel_launch(void* const* d_in, const int* in_sizes, int n_in,
                              void* d_out, int out_size, void* d_ws, size_t ws_size,
                              hipStream_t stream) {
  (void)in_sizes; (void)n_in; (void)out_size; (void)ws_size;
  const float* x_f     = (const float*)d_in[0];
  const float* x_p     = (const float*)d_in[1];
  const float* other   = (const float*)d_in[2];
  const int*   ei      = (const int*)d_in[3];
  const float* pnode_w = (const float*)d_in[4];
  const float* pnode_b = (const float*)d_in[5];
  const float* fnode_w = (const float*)d_in[6];
  const float* fnode_b = (const float*)d_in[7];
  const float* c1wl[3] = {(const float*)d_in[8],  (const float*)d_in[11], (const float*)d_in[14]};
  const float* c1bl[3] = {(const float*)d_in[9],  (const float*)d_in[12], (const float*)d_in[15]};
  const float* c1wr[3] = {(const float*)d_in[10], (const float*)d_in[13], (const float*)d_in[16]};
  const float* c2wl[3] = {(const float*)d_in[17], (const float*)d_in[20], (const float*)d_in[23]};
  const float* c2bl[3] = {(const float*)d_in[18], (const float*)d_in[21], (const float*)d_in[24]};
  const float* c2wr[3] = {(const float*)d_in[19], (const float*)d_in[22], (const float*)d_in[25]};
  const float* lnw[2]  = {(const float*)d_in[26], (const float*)d_in[28]};
  const float* lnb[2]  = {(const float*)d_in[27], (const float*)d_in[29]};
  const float* conv_w[3] = {(const float*)d_in[30], (const float*)d_in[32], (const float*)d_in[34]};
  const float* conv_b[3] = {(const float*)d_in[31], (const float*)d_in[33], (const float*)d_in[35]};
  const float* d1w = (const float*)d_in[36];
  const float* d1b = (const float*)d_in[37];
  const float* d2w = (const float*)d_in[38];
  const float* d2b = (const float*)d_in[39];
  const float* d3w = (const float*)d_in[40];
  const float* d3b = (const float*)d_in[41];
  float* out = (float*)d_out;

  // ---- workspace layout (float units) ----
  float* ws = (float*)d_ws;
  float* xp_a  = ws;                       // 32768*128
  float* xp_b  = xp_a + 4194304;
  float* xf_a  = xp_b + 4194304;           // 6144*128
  float* xf_b  = xf_a + 786432;
  float* agg   = xf_b + 786432;            // 32768*128
  float* cnt_f = agg + 4194304;            // 32768
  float* cnt_b = cnt_f + 32768;            // 6144
  float* c1out = cnt_b + 6144;             // 8*64*4089
  float* c2out = c1out + 2093568;          // 8*64*4082
  float* c3out = c2out + 2089984;          // 8*64*4075
  float* d1out = c3out + 2086400;          // 8*100
  __bf16* pk   = (__bf16*)(d1out + 800);   // packed weight images (bf16)
  __bf16* pk_c1wl[3], *pk_c1wr[3], *pk_c2wl[3], *pk_c2wr[3];
  for (int i = 0; i < 3; ++i) {
    pk_c1wl[i] = pk + (0 + i) * 16384;
    pk_c1wr[i] = pk + (3 + i) * 16384;
    pk_c2wl[i] = pk + (6 + i) * 16384;
    pk_c2wr[i] = pk + (9 + i) * 16384;
  }
  __bf16* pk_cv1 = pk + 12 * 16384;        // 1024*64
  __bf16* pk_cv2 = pk_cv1 + 65536;         // 512*64
  __bf16* pk_cv3 = pk_cv2 + 32768;         // 512*64

  const int T = 256;
  auto blks = [](int n, int t) { return (n + t - 1) / t; };
  auto packB = [&](const float* W, __bf16* P, int K, int Kr, int NT, int sk, int sn) {
    int total = (Kr / 32) * NT * 512;
    pack_wmma_b<<<blks(total, T), T, 0, stream>>>(W, P, K, NT, sk, sn, total);
  };

  // ---- per-call init (graph-replay safe) ----
  zero_kernel<<<blks(PN_ROWS * 128, T), T, 0, stream>>>(xp_a, PN_ROWS * 128);
  zero_kernel<<<blks(FN_ROWS * 128, T), T, 0, stream>>>(xf_a, FN_ROWS * 128);
  zero_kernel<<<blks(32768, T), T, 0, stream>>>(cnt_f, 32768);
  zero_kernel<<<blks(6144, T), T, 0, stream>>>(cnt_b, 6144);
  zero_kernel<<<blks(800, T), T, 0, stream>>>(d1out, 800);
  count_edges<<<blks(NE_HALF, T), T, 0, stream>>>(ei, cnt_f, NE_HALF, 1);
  count_edges<<<blks(NE_HALF, T), T, 0, stream>>>(ei, cnt_b, NE_HALF, 0);

  // ---- pack all WMMA B-operands (bf16, fragment order, K zero-padded) ----
  packB(c1wl[0], pk_c1wl[0], 3, 32, 8, 128, 1);
  packB(c1wr[0], pk_c1wr[0], 3, 32, 8, 128, 1);
  packB(c2wl[0], pk_c2wl[0], 128, 128, 8, 128, 1);
  packB(c2wr[0], pk_c2wr[0], 3, 32, 8, 128, 1);
  for (int i = 1; i < 3; ++i) {
    packB(c1wl[i], pk_c1wl[i], 128, 128, 8, 128, 1);
    packB(c1wr[i], pk_c1wr[i], 128, 128, 8, 128, 1);
    packB(c2wl[i], pk_c2wl[i], 128, 128, 8, 128, 1);
    packB(c2wr[i], pk_c2wr[i], 128, 128, 8, 128, 1);
  }
  packB(conv_w[0], pk_cv1, 1024, 1024, 4, 1, 1024);  // w[n*K + k]
  packB(conv_w[1], pk_cv2, 512, 512, 4, 1, 512);
  packB(conv_w[2], pk_cv3, 512, 512, 4, 1, 512);

  // ---- node encoders ----
  pnode_gemm<<<96, 128, 0, stream>>>(x_p, pnode_w, pnode_b, xp_a);
  fnode_gemm<<<blks(2048 * 192, T), T, 0, stream>>>(x_f, fnode_w, fnode_b, xf_a);

  float *xp_c = xp_a, *xp_n = xp_b, *xf_c = xf_a, *xf_n = xf_b;
  int df = 3, dp = 3;
  for (int i = 0; i < 3; ++i) {
    int ksF = (df == 3) ? 1 : 4, ksP = (dp == 3) ? 1 : 4;
    // forward: aggregate fragment feats into protein nodes
    zero_kernel<<<blks(PN_ROWS * 128, T), T, 0, stream>>>(agg, PN_ROWS * 128);
    scatter_add<<<blks(NE_HALF * df, T), T, 0, stream>>>(ei, xf_c, agg, NE_HALF, df, 1);
    sage_update<<<PN_ROWS / 128, T, 0, stream>>>(
        agg, cnt_f, xp_c, (const v16bf*)pk_c1wl[i], c1bl[i], (const v16bf*)pk_c1wr[i],
        xp_n, ksF, ksP);
    { float* t = xp_c; xp_c = xp_n; xp_n = t; } dp = 128;
    // backward: aggregate protein feats into fragment nodes
    zero_kernel<<<blks(FN_ROWS * 128, T), T, 0, stream>>>(agg, FN_ROWS * 128);
    scatter_add<<<blks(NE_HALF * 128, T), T, 0, stream>>>(ei, xp_c, agg, NE_HALF, 128, 0);
    sage_update<<<FN_ROWS / 128, T, 0, stream>>>(
        agg, cnt_b, xf_c, (const v16bf*)pk_c2wl[i], c2bl[i], (const v16bf*)pk_c2wr[i],
        xf_n, 4, ksF);
    { float* t = xf_c; xf_c = xf_n; xf_n = t; } df = 128;
    if (i < 2) {
      ln_kernel<<<blks(PN_ROWS, 8), T, 0, stream>>>(xp_c, lnw[i], lnb[i], PN_ROWS);
      ln_kernel<<<blks(FN_ROWS, 8), T, 0, stream>>>(xf_c, lnw[i], lnb[i], FN_ROWS);
    }
  }

  // ---- conv chain: xp_c contiguous (32768,128) == (8,128,4096) ----
  conv1d_relu_wmma<<<dim3(blks(4089, 128), 8), T, 0, stream>>>(
      xp_c, (const v16bf*)pk_cv1, conv_b[0], c1out, 128, 4096, 4089);
  conv1d_relu_wmma<<<dim3(blks(4082, 128), 8), T, 0, stream>>>(
      c1out, (const v16bf*)pk_cv2, conv_b[1], c2out, 64, 4089, 4082);
  conv1d_relu_wmma<<<dim3(blks(4075, 128), 8), T, 0, stream>>>(
      c2out, (const v16bf*)pk_cv3, conv_b[2], c3out, 64, 4082, 4075);

  // ---- d1 (HBM-bound over 104MB weights) + tail ----
  d1_partial<<<blks(260800, 512), 128, 0, stream>>>(c3out, d1w, d1out);
  tail_kernel<<<1, 256, 0, stream>>>(d1out, d1b, other, d2w, d2b, d3w, d3b, out);
}
